// Attention_67044439491254
// MI455X (gfx1250) — matrix operand order repaired and checked by reading, not compiled, not run on hardware
//
#include <hip/hip_runtime.h>
#include <hip/hip_bf16.h>

// ---------------------------------------------------------------------------
// Problem constants (from reference): B=2, N=2048, C=256, hidden=1024, H=16, d=64
// ---------------------------------------------------------------------------
constexpr int Bb   = 2;
constexpr int Nn   = 2048;
constexpr int Cc   = 256;
constexpr int HID  = 1024;   // 4*C
constexpr int Hh   = 16;
constexpr int Dd   = 64;     // head dim
constexpr int MR   = Bb * Nn;     // 4096 rows of activations
constexpr int QKVC = 3 * HID;     // 3072

typedef __attribute__((ext_vector_type(16))) __bf16        bf16x16;
typedef __attribute__((ext_vector_type(8)))  float         f32x8;
typedef __attribute__((ext_vector_type(8)))  unsigned int  u32x8;
typedef __attribute__((ext_vector_type(4)))  unsigned int  tdm_u32x4;
typedef __attribute__((ext_vector_type(8)))  int           tdm_i32x8;
typedef __attribute__((ext_vector_type(4)))  int           tdm_i32x4;

// float -> bf16 (round to nearest even)
__device__ __forceinline__ unsigned short f2bf(float f) {
    unsigned int u = __builtin_bit_cast(unsigned int, f);
    u += 0x7FFFu + ((u >> 16) & 1u);
    return (unsigned short)(u >> 16);
}

__device__ __forceinline__ f32x8 wmma_bf16(bf16x16 a, bf16x16 b, f32x8 c) {
    return __builtin_amdgcn_wmma_f32_16x16x32_bf16(
        /*neg_a=*/false, a, /*neg_b=*/false, b,
        /*c_mod=*/(short)0, c, /*reuse_a=*/false, /*reuse_b=*/false);
}

// A-fragment: 16x32 bf16, element (m,k) at A[m*lda + k] (row-major LDS tile).
// ISA layout: lanes 0-15 -> M=lane, K in {0..7,16..23}; lanes 16-31 -> K in {8..15,24..31}.
__device__ __forceinline__ bf16x16 ld_frag_a(const unsigned short* A, int lda, int lane) {
    const unsigned short* row = A + (lane & 15) * lda;
    const int kh = (lane & 16) ? 8 : 0;
    u32x8 f;
#pragma unroll
    for (int p = 0; p < 8; ++p) {
        const int k = ((p & 4) ? 16 : 0) + kh + ((p & 3) << 1);
        f[p] = *reinterpret_cast<const unsigned int*>(row + k);
    }
    return __builtin_bit_cast(bf16x16, f);
}

// B-fragment: 32x16 bf16 from a K-transposed LDS tile Bt where Bt[n*ldb + k] = B[k][n].
// ISA layout: lanes 0-15 -> N=lane, K in {0..7,16..23}; lanes 16-31 -> K in {8..15,24..31} (2 per VGPR).
__device__ __forceinline__ bf16x16 ld_frag_b(const unsigned short* Bt, int ldb, int lane) {
    const unsigned short* row = Bt + (lane & 15) * ldb + ((lane & 16) ? 8 : 0);
    u32x8 f;
#pragma unroll
    for (int p = 0; p < 8; ++p)
        f[p] = *reinterpret_cast<const unsigned int*>(row + ((p & 4) ? 16 : 0) + ((p & 3) << 1));
    return __builtin_bit_cast(bf16x16, f);
}

// ---------------------------------------------------------------------------
// Tensor Data Mover: 2-D bf16 tile (tile_rows x 64) from global -> LDS with
// LDS row padding 64 -> 72 ushorts (pad_interval = 32 dwords, pad_amount = 4
// dwords).  Descriptor per CDNA5 ISA; groups 2/3 unused
// for 2-D tensors.  Tracked by TENSORcnt.  (6-arg builtin form: clang-23.)
// ---------------------------------------------------------------------------
__device__ __forceinline__ void tdm_load_tile64(unsigned lds_off, const void* gsrc,
                                                unsigned tile_rows) {
    const unsigned long long ga = (unsigned long long)gsrc;
    tdm_u32x4 g0;
    g0[0] = 1u;                                   // count=1, user descriptor
    g0[1] = lds_off;                              // LDS byte address
    g0[2] = (unsigned)ga;                         // global_addr[31:0]
    g0[3] = (unsigned)((ga >> 32) & 0x01FFFFFFu)  // global_addr[56:32]
          | (2u << 30);                           // type = 2 ("image")
    tdm_i32x8 g1;
    g1[0] = (int)((1u << 16)      // data_size = 2 bytes (bf16)
          |       (1u << 20)      // pad_enable
          |       (4u << 22)      // pad_interval: 2^(4+1)=32 dwords (one 128B row)
          |       (3u << 25));    // pad_amount: 3+1 = 4 dwords (16B) -> stride 72
    g1[1] = (int)(64u << 16);     // tensor_dim0[15:0]=64 in bits[63:48]; barrier addr 0
    g1[2] = (int)(2048u << 16);   // tensor_dim1[15:0]=2048 in bits[95:80]
    g1[3] = (int)(64u << 16);     // tile_dim0 = 64 in bits[127:112]
    g1[4] = (int)tile_rows;       // tile_dim1 = rows; tile_dim2 = 0
    g1[5] = 64;                   // tensor_dim0_stride = 64 elements
    g1[6] = 0;
    g1[7] = 0;
    tdm_i32x4 z4 = {0, 0, 0, 0};
    tdm_i32x8 z8 = {0, 0, 0, 0, 0, 0, 0, 0};
    __builtin_amdgcn_tensor_load_to_lds(g0, g1, z4, z4, z8, 0);
}

// ---------------------------------------------------------------------------
// Kernel 1: qkv = array @ Wqkv  (fp32 in, bf16 out, scattered to (B*H, N, 64))
// Block: 256 threads (8 waves). Tile: 64 rows x 128 cols, K-step 32.
// (Staging needs fp32->bf16 conversion + B transpose, so VALU staging, not TDM.)
// ---------------------------------------------------------------------------
__global__ __launch_bounds__(256) void qkv_gemm(const float* __restrict__ A,
                                                const float* __restrict__ W,
                                                unsigned short* __restrict__ qo,
                                                unsigned short* __restrict__ ko,
                                                unsigned short* __restrict__ vo) {
    __shared__ alignas(16) unsigned short Ash[64 * 40];
    __shared__ alignas(16) unsigned short Bts[128 * 40];
    const int t    = threadIdx.x;
    const int lane = t & 31;
    const int wv   = t >> 5;
    const int bm   = blockIdx.x;   // 64-row tile
    const int bn   = blockIdx.y;   // 128-col tile
    const int mrow = (wv & 3) * 16;
    const int ncol = (wv >> 2) * 64;

    f32x8 acc[4];
#pragma unroll
    for (int i = 0; i < 4; ++i) acc[i] = (f32x8){0.f,0.f,0.f,0.f,0.f,0.f,0.f,0.f};

    for (int k0 = 0; k0 < Cc; k0 += 32) {
#pragma unroll
        for (int j = 0; j < 8; ++j) {
            const int idx = j * 256 + t;
            const int c = idx & 31, r = idx >> 5;
            Ash[r * 40 + c] = f2bf(A[(bm * 64 + r) * Cc + k0 + c]);
        }
#pragma unroll
        for (int j = 0; j < 16; ++j) {
            const int idx = j * 256 + t;
            const int n = idx & 127, kk = idx >> 7;
            Bts[n * 40 + kk] = f2bf(W[(k0 + kk) * QKVC + bn * 128 + n]);
        }
        __syncthreads();

        const bf16x16 af = ld_frag_a(Ash + mrow * 40, 40, lane);
#pragma unroll
        for (int nt = 0; nt < 4; ++nt) {
            const bf16x16 bf = ld_frag_b(Bts + (ncol + nt * 16) * 40, 40, lane);
            acc[nt] = wmma_bf16(af, bf, acc[nt]);
        }
        __syncthreads();
    }

    const int nloc = lane & 15;
    const int half = lane >> 4;
#pragma unroll
    for (int nt = 0; nt < 4; ++nt) {
        const int col    = bn * 128 + ncol + nt * 16 + nloc;
        const int part   = col >> 10;
        const int within = col & (HID - 1);
        const int h      = within >> 6;
        const int dd     = within & 63;
        unsigned short* dst = (part == 0) ? qo : ((part == 1) ? ko : vo);
#pragma unroll
        for (int r = 0; r < 8; ++r) {
            const int row = bm * 64 + mrow + r + half * 8;
            const int b   = row >> 11;
            const int n   = row & (Nn - 1);
            dst[(((b * Hh + h) * Nn + n) << 6) + dd] = f2bf(acc[nt][r]);
        }
    }
}

// ---------------------------------------------------------------------------
// Kernel 2: flash attention per (b,h).  Block: 128 threads (4 waves), 64 Q rows.
// Q and K tiles staged LDS-side by the Tensor Data Mover (bf16, row-major,
// padded stride 72); V staged transposed by the VALU path (TDM can't transpose).
// ---------------------------------------------------------------------------
__global__ __launch_bounds__(128) void flash_attn(const unsigned short* __restrict__ Q,
                                                  const unsigned short* __restrict__ K,
                                                  const unsigned short* __restrict__ V,
                                                  unsigned short* __restrict__ O) {
    __shared__ alignas(16) unsigned short Qsh[64 * 72];
    __shared__ alignas(16) unsigned short Ksh[32 * 72];     // row-major == Bt for Q*K^T
    __shared__ alignas(16) unsigned short Vt [64 * 40];     // Vt[dd][j] = V[j][dd]
    __shared__ alignas(16) unsigned short Psh[4 * 16 * 40]; // per-wave P staging

    const int t    = threadIdx.x;
    const int lane = t & 31;
    const int wv   = t >> 5;
    const int bh   = blockIdx.x;          // b*16 + h
    const int row0 = blockIdx.y * 64;

    const unsigned short* Qb = Q + (size_t)bh * Nn * Dd;
    const unsigned short* Kb = K + (size_t)bh * Nn * Dd;
    const unsigned short* Vb = V + (size_t)bh * Nn * Dd;

    // LDS offsets for TDM descriptors: low 32 bits of the generic shared-mem
    // address are the workgroup-relative LDS byte address (ISA §10.2 aperture).
    const unsigned qsh_off = (unsigned)(unsigned long long)(const void*)&Qsh[0];
    const unsigned ksh_off = (unsigned)(unsigned long long)(const void*)&Ksh[0];

    // Stage Q tile (64x64 bf16 -> 64x72 padded) via the Tensor Data Mover.
    if (wv == 0) {
        tdm_load_tile64(qsh_off, Qb + (size_t)row0 * Dd, 64u);
        __builtin_amdgcn_s_wait_tensorcnt(0);
    }
    __syncthreads();
    const bf16x16 qa0 = ld_frag_a(Qsh + wv * 16 * 72,      72, lane);
    const bf16x16 qa1 = ld_frag_a(Qsh + wv * 16 * 72 + 32, 72, lane);

    f32x8 o0 = (f32x8){0.f,0.f,0.f,0.f,0.f,0.f,0.f,0.f};
    f32x8 o1 = o0, o2 = o0, o3 = o0;
    float m_i[8], l_i[8];
#pragma unroll
    for (int r = 0; r < 8; ++r) { m_i[r] = -3.0e38f; l_i[r] = 0.f; }

    for (int kv = 0; kv < Nn; kv += 32) {
        // K tile (32x64 row-major) via TDM; overlaps with V transpose staging.
        if (wv == 0)
            tdm_load_tile64(ksh_off, Kb + (size_t)kv * Dd, 32u);
#pragma unroll
        for (int j = 0; j < 16; ++j) {
            const int idx = j * 128 + t;
            const int c = idx & 63, r = idx >> 6;
            Vt[c * 40 + r] = Vb[(kv + r) * Dd + c];
        }
        if (wv == 0)
            __builtin_amdgcn_s_wait_tensorcnt(0);
        __syncthreads();

        // S(16x32) = Q(16x64) * K^T : two n-tiles, K-dim (d) in two steps of 32
        f32x8 s0 = (f32x8){0.f,0.f,0.f,0.f,0.f,0.f,0.f,0.f};
        f32x8 s1 = s0;
        s0 = wmma_bf16(qa0, ld_frag_b(Ksh,                72, lane), s0);
        s0 = wmma_bf16(qa1, ld_frag_b(Ksh + 32,           72, lane), s0);
        s1 = wmma_bf16(qa0, ld_frag_b(Ksh + 16 * 72,      72, lane), s1);
        s1 = wmma_bf16(qa1, ld_frag_b(Ksh + 16 * 72 + 32, 72, lane), s1);
        s0 *= 0.125f;   // 1/sqrt(64)
        s1 *= 0.125f;

        // Online softmax: each lane owns column (lane&15) of rows r + 8*(lane>>4)
#pragma unroll
        for (int r = 0; r < 8; ++r) {
            float tmax = fmaxf(s0[r], s1[r]);
            tmax = fmaxf(tmax, __shfl_xor(tmax, 1));
            tmax = fmaxf(tmax, __shfl_xor(tmax, 2));
            tmax = fmaxf(tmax, __shfl_xor(tmax, 4));
            tmax = fmaxf(tmax, __shfl_xor(tmax, 8));
            const float mnew  = fmaxf(m_i[r], tmax);
            const float alpha = __expf(m_i[r] - mnew);
            const float p0 = __expf(s0[r] - mnew);
            const float p1 = __expf(s1[r] - mnew);
            s0[r] = p0; s1[r] = p1;
            float rs = p0 + p1;
            rs += __shfl_xor(rs, 1);
            rs += __shfl_xor(rs, 2);
            rs += __shfl_xor(rs, 4);
            rs += __shfl_xor(rs, 8);
            l_i[r] = l_i[r] * alpha + rs;
            m_i[r] = mnew;
            o0[r] *= alpha; o1[r] *= alpha; o2[r] *= alpha; o3[r] *= alpha;
        }

        // D-layout -> A-layout for P via per-wave LDS round trip
        unsigned short* Pw = Psh + wv * 16 * 40;
#pragma unroll
        for (int r = 0; r < 8; ++r) {
            const int m = r + (lane >> 4) * 8;
            Pw[m * 40 +      (lane & 15)] = f2bf(s0[r]);
            Pw[m * 40 + 16 + (lane & 15)] = f2bf(s1[r]);
        }
        // Same-wave cross-lane LDS RAW: LDS is in-order per wave; make the
        // ordering explicit with the CDNA5 split DS counter wait.
        asm volatile("s_wait_dscnt 0" ::: "memory");
        const bf16x16 pf = ld_frag_a(Pw, 40, lane);

        // O(16x64) += P(16x32) * V(32x64)
        o0 = wmma_bf16(pf, ld_frag_b(Vt,           40, lane), o0);
        o1 = wmma_bf16(pf, ld_frag_b(Vt + 16 * 40, 40, lane), o1);
        o2 = wmma_bf16(pf, ld_frag_b(Vt + 32 * 40, 40, lane), o2);
        o3 = wmma_bf16(pf, ld_frag_b(Vt + 48 * 40, 40, lane), o3);
        __syncthreads();   // everyone done with Ksh/Vt before next staging
    }

    // Normalize and store to (B, N, hidden) bf16 for the output projection
    const int b = bh >> 4, h = bh & 15;
#pragma unroll
    for (int r = 0; r < 8; ++r) {
        const float inv = 1.0f / l_i[r];
        const int n = row0 + wv * 16 + r + (lane >> 4) * 8;
        unsigned short* dst = O + (size_t)(b * Nn + n) * HID + h * 64 + (lane & 15);
        dst[0]  = f2bf(o0[r] * inv);
        dst[16] = f2bf(o1[r] * inv);
        dst[32] = f2bf(o2[r] * inv);
        dst[48] = f2bf(o3[r] * inv);
    }
}

// ---------------------------------------------------------------------------
// Kernel 3: out = attn_out(4096x1024, bf16) @ Wout(1024x64, fp32) -> fp32
// Block: 256 threads (8 waves), tile 128 rows x 64 cols, K-step 32.
// ---------------------------------------------------------------------------
__global__ __launch_bounds__(256) void out_gemm(const unsigned short* __restrict__ A,
                                                const float* __restrict__ W,
                                                float* __restrict__ out) {
    __shared__ alignas(16) unsigned short Ash[128 * 40];
    __shared__ alignas(16) unsigned short Bts[64 * 40];
    const int t    = threadIdx.x;
    const int lane = t & 31;
    const int wv   = t >> 5;
    const int bm   = blockIdx.x;
    const int mrow = wv * 16;

    f32x8 acc[4];
#pragma unroll
    for (int i = 0; i < 4; ++i) acc[i] = (f32x8){0.f,0.f,0.f,0.f,0.f,0.f,0.f,0.f};

    for (int k0 = 0; k0 < HID; k0 += 32) {
#pragma unroll
        for (int j = 0; j < 16; ++j) {
            const int idx = j * 256 + t;
            const int c = idx & 31, r = idx >> 5;
            Ash[r * 40 + c] = A[(bm * 128 + r) * HID + k0 + c];
        }
#pragma unroll
        for (int j = 0; j < 8; ++j) {
            const int idx = j * 256 + t;
            const int n = idx & 63, kk = idx >> 6;
            Bts[n * 40 + kk] = f2bf(W[(k0 + kk) * Dd + n]);
        }
        __syncthreads();

        const bf16x16 af = ld_frag_a(Ash + mrow * 40, 40, lane);
#pragma unroll
        for (int nt = 0; nt < 4; ++nt)
            acc[nt] = wmma_bf16(af, ld_frag_b(Bts + nt * 16 * 40, 40, lane), acc[nt]);
        __syncthreads();
    }

    const int nloc = lane & 15;
    const int half = lane >> 4;
#pragma unroll
    for (int nt = 0; nt < 4; ++nt)
#pragma unroll
        for (int r = 0; r < 8; ++r)
            out[(bm * 128 + mrow + r + half * 8) * Dd + nt * 16 + nloc] = acc[nt][r];
}

// ---------------------------------------------------------------------------
// Host launcher
// ---------------------------------------------------------------------------
extern "C" void kernel_launch(void* const* d_in, const int* in_sizes, int n_in,
                              void* d_out, int out_size, void* d_ws, size_t ws_size,
                              hipStream_t stream) {
    (void)in_sizes; (void)n_in; (void)out_size; (void)ws_size;
    const float* arr  = (const float*)d_in[0];   // (2, 2048, 256)
    const float* Wqkv = (const float*)d_in[1];   // (256, 3072)
    const float* Wout = (const float*)d_in[2];   // (1024, 64)
    float* out = (float*)d_out;                  // (2, 2048, 64)

    const size_t QKV_ELEMS = (size_t)Bb * Hh * Nn * Dd;  // 4,194,304
    unsigned short* q  = (unsigned short*)d_ws;
    unsigned short* k  = q + QKV_ELEMS;
    unsigned short* v  = k + QKV_ELEMS;
    unsigned short* oB = v + QKV_ELEMS;

    dim3 g1(MR / 64, QKVC / 128);                  // 64 x 24
    qkv_gemm<<<g1, 256, 0, stream>>>(arr, Wqkv, q, k, v);

    dim3 g2(Bb * Hh, Nn / 64);                     // 32 x 32
    flash_attn<<<g2, 128, 0, stream>>>(q, k, v, oB);

    out_gemm<<<MR / 128, 256, 0, stream>>>(oB, Wout, out);
}
